// TransformerBlock_52931176956092
// MI455X (gfx1250) — matrix operand order repaired and checked by reading, not compiled, not run on hardware
//
#include <hip/hip_runtime.h>
#include <math.h>

// ---------------- problem constants ----------------
#define D_MODEL    1024
#define N_HEADS    16
#define HEAD_DIM   64
#define MLP_HIDDEN 4096
#define SEQ_B      8
#define SEQ_L      1024
#define TOKENS     (SEQ_B * SEQ_L)   // 8192

// ---------------- WMMA / TDM vector types ----------------
typedef __attribute__((ext_vector_type(16))) __bf16 v16bf;
typedef __attribute__((ext_vector_type(8)))  float  v8f;
typedef __attribute__((ext_vector_type(4)))  unsigned int u32x4;
typedef __attribute__((ext_vector_type(8)))  int  i32x8;
typedef __attribute__((ext_vector_type(4)))  int  i32x4;

// fp32 -> bf16 (RNE via native conversion)
__device__ __forceinline__ unsigned short bf(float f) {
  __bf16 h = (__bf16)f;
  return __builtin_bit_cast(unsigned short, h);
}
__device__ __forceinline__ unsigned pack_bf2(float a, float b) {
  return (unsigned)bf(a) | ((unsigned)bf(b) << 16);
}

// B-operand fill: 16 consecutive bf16 (K = kbase .. kbase+15), lane holds col N
__device__ __forceinline__ v16bf load_bop(const unsigned short* p) {
  union { v16bf v; uint4 q[2]; } u;
  u.q[0] = *(const uint4*)(p);
  u.q[1] = *(const uint4*)(p + 8);
  return u.v;
}
// A-operand fill: K = c0..c0+15 (contiguous), lane holds row M
__device__ __forceinline__ v16bf load_aop(const unsigned short* row, int c0) {
  union { v16bf v; uint4 q[2]; } u;
  u.q[0] = *(const uint4*)(row + c0);
  u.q[1] = *(const uint4*)(row + c0 + 8);
  return u.v;
}

// ---------------- Tensor Data Mover: 2D bf16 tile -> LDS ----------------
// D# per CDNA5 ISA ch.8: group0 = {flags, lds_addr, global_addr, type=2},
// group1 = dims/strides. pad_code: LDS padding interval code
//   3 -> pad every 16 DWORDs (32 bf16 row), 4 -> every 32 DWORDs (64 bf16 row);
// pad amount code 3 -> 4 DWORDs (16B = 8 bf16) => padded strides 40 / 72.
__device__ __forceinline__ void tdm_load_2d(unsigned lds_byte_off, const void* gaddr,
                                            unsigned tile0, unsigned tile1,
                                            unsigned td0, unsigned long stride0,
                                            int pad_code)
{
  unsigned long ga = (unsigned long)gaddr;
  const unsigned td1 = 1u << 30;
  u32x4 g0;
  g0[0] = 1u;                                  // count=1 (valid user descriptor)
  g0[1] = lds_byte_off;                        // lds_addr
  g0[2] = (unsigned)ga;                        // global_addr[31:0]
  g0[3] = (unsigned)(ga >> 32) | (2u << 30);   // global_addr[56:32] | type=2
  i32x8 g1;
  g1[0] = (int)((1u << 16) | (1u << 20) | ((unsigned)pad_code << 22) | (3u << 25));
      // data_size=1 (2B), pad_enable, pad_interval, pad_amount=3 (4 DWORDs)
  g1[1] = (int)(td0 << 16);                            // tensor_dim0[15:0]
  g1[2] = (int)((td0 >> 16) | (td1 << 16));            // dim0[31:16] | dim1[15:0]
  g1[3] = (int)((td1 >> 16) | (tile0 << 16));          // dim1[31:16] | tile_dim0
  g1[4] = (int)(tile1 & 0xFFFFu);                      // tile_dim1 (tile_dim2=0)
  g1[5] = (int)(unsigned)stride0;                      // tensor_dim0_stride[31:0]
  g1[6] = (int)((unsigned)(stride0 >> 32) & 0xFFFFu);  // stride0[47:32]
  g1[7] = 0;
  i32x4 gz4 = {0, 0, 0, 0};
  i32x8 gz8 = {0, 0, 0, 0, 0, 0, 0, 0};
  // 6-arg form (clang-23 / therock-10.0 headers): groups 2,3 unused (2D tile)
  __builtin_amdgcn_tensor_load_to_lds(g0, g1, gz4, gz4, gz8, 0);
}

// ================= weight convert + transpose: Wt[n,k] = bf16(W[k,n]) =======
__global__ __launch_bounds__(256) void wt_transpose_bf16(
    const float* __restrict__ W, unsigned short* __restrict__ Wt, int K, int N)
{
  __shared__ float T[64][65];
  const int k0 = blockIdx.x * 64;
  const int n0 = blockIdx.y * 64;
  const int tid = threadIdx.x;
  #pragma unroll
  for (int i = 0; i < 16; ++i) {
    int idx = tid + i * 256;
    int kr = idx >> 6, nc = idx & 63;
    T[kr][nc] = W[(size_t)(k0 + kr) * N + n0 + nc];
  }
  __syncthreads();
  #pragma unroll
  for (int i = 0; i < 16; ++i) {
    int idx = tid + i * 256;
    int nr = idx >> 6, kc = idx & 63;
    Wt[(size_t)(n0 + nr) * K + k0 + kc] = bf(T[kc][nr]);
  }
}

// ================= LayerNorm -> bf16 (one block per row, C = 1024) ==========
__global__ __launch_bounds__(256) void ln_bf16_kernel(
    const float* __restrict__ x, const float* __restrict__ g,
    const float* __restrict__ beta, unsigned short* __restrict__ y)
{
  const int row = blockIdx.x;
  const int tid = threadIdx.x;
  const float4 v = *(const float4*)(x + (size_t)row * D_MODEL + tid * 4);
  float s  = v.x + v.y + v.z + v.w;
  float s2 = v.x*v.x + v.y*v.y + v.z*v.z + v.w*v.w;
  #pragma unroll
  for (int off = 16; off >= 1; off >>= 1) {
    s  += __shfl_xor(s,  off, 32);
    s2 += __shfl_xor(s2, off, 32);
  }
  __shared__ float w1[8], w2[8];
  if ((tid & 31) == 0) { w1[tid >> 5] = s; w2[tid >> 5] = s2; }
  __syncthreads();
  s = 0.f; s2 = 0.f;
  #pragma unroll
  for (int i = 0; i < 8; ++i) { s += w1[i]; s2 += w2[i]; }
  const float mu   = s  * (1.0f / D_MODEL);
  const float var  = s2 * (1.0f / D_MODEL) - mu * mu;
  const float rinv = rsqrtf(var + 1e-5f);
  const int c = tid * 4;
  const float o0 = (v.x - mu) * rinv * g[c+0] + beta[c+0];
  const float o1 = (v.y - mu) * rinv * g[c+1] + beta[c+1];
  const float o2 = (v.z - mu) * rinv * g[c+2] + beta[c+2];
  const float o3 = (v.w - mu) * rinv * g[c+3] + beta[c+3];
  uint2 o; o.x = pack_bf2(o0, o1); o.y = pack_bf2(o2, o3);
  *(uint2*)(y + (size_t)row * D_MODEL + c) = o;
}

// ================= bf16 WMMA GEMM (TDM-staged, double buffered) =============
// out[M,N] = A[M,K](bf16) @ Wt[N,K]^T(bf16) + epilogue
#define BM 64
#define BN 128
#define BK 32
// LDS layout per buffer (halfword units): As 64 rows x 40, Bs 128 rows x 40
#define AS_STRIDE 40
#define BUF_HALFS (64 * AS_STRIDE + 128 * AS_STRIDE)     // 7680
#define BUF_BYTES (BUF_HALFS * 2)                         // 15360
#define BS_BYTE_OFF (64 * AS_STRIDE * 2)                  // 5120

enum { OP_BIAS = 0, OP_BIAS_RES = 1, OP_BIAS_GELU = 2 };

template <int OP, int OUTBF>
__global__ __launch_bounds__(256) void gemm_bf16_tdm(
    const unsigned short* __restrict__ A, const unsigned short* __restrict__ Wt,
    const float* __restrict__ bias, const float* __restrict__ res,
    void* __restrict__ outp, int M, int N, int K)
{
  __shared__ __align__(16) unsigned short sm[2 * BUF_HALFS];

  const int tid  = threadIdx.x;
  const int lane = tid & 31;
  const int wave = tid >> 5;
  const int wm   = wave & 3;        // 4 waves x 16 rows
  const int wn   = wave >> 2;       // 2 waves x 64 cols (4 n-tiles each)
  const int m0   = blockIdx.y * BM;
  const int n0   = blockIdx.x * BN;
  const int lm   = lane & 15;
  const int hi   = lane >> 4;
  const int c0   = hi ? 16 : 0;     // A-operand K start
  const int kb16 = hi ? 16 : 0;     // B-operand K start

  v8f acc[4];
  #pragma unroll
  for (int t = 0; t < 4; ++t) acc[t] = (v8f){0.f,0.f,0.f,0.f,0.f,0.f,0.f,0.f};

  const int nk = K / BK;
  auto issue_tiles = [&](int step, int buf) {
    const unsigned short* ga = A  + (size_t)m0 * K + step * BK;
    tdm_load_2d((unsigned)(buf * BUF_BYTES), ga, BK, BM, (unsigned)K,
                (unsigned long)K, 3);
    const unsigned short* gb = Wt + (size_t)n0 * K + step * BK;
    tdm_load_2d((unsigned)(buf * BUF_BYTES + BS_BYTE_OFF), gb, BK, BN,
                (unsigned)K, (unsigned long)K, 3);
  };

  if (wave == 0) issue_tiles(0, 0);

  int buf = 0;
  for (int i = 0; i < nk; ++i) {
    __syncthreads();                 // prior consumers of buf^1 are done
    if (wave == 0) {
      if (i + 1 < nk) {
        issue_tiles(i + 1, buf ^ 1);
        __builtin_amdgcn_s_wait_tensorcnt(2);   // current buffer complete
      } else {
        __builtin_amdgcn_s_wait_tensorcnt(0);
      }
    }
    __syncthreads();                 // publish buf to all waves

    const unsigned short* as = &sm[buf * BUF_HALFS];
    const unsigned short* bs = as + 64 * AS_STRIDE;
    v16bf av = load_aop(as + (wm * 16 + lm) * AS_STRIDE, c0);
    #pragma unroll
    for (int t = 0; t < 4; ++t) {
      v16bf bv = load_bop(bs + (wn * 64 + t * 16 + lm) * AS_STRIDE + kb16);
      acc[t] = __builtin_amdgcn_wmma_f32_16x16x32_bf16(false, av, false, bv,
                                                       (short)0, acc[t], false, false);
    }
    buf ^= 1;
  }

  // epilogue: D layout -> lane holds col n=lm, rows r (+8 for upper half-wave)
  const int rbase = hi * 8;
  #pragma unroll
  for (int t = 0; t < 4; ++t) {
    const int n = n0 + wn * 64 + t * 16 + lm;
    const float bv = bias[n];
    #pragma unroll
    for (int r = 0; r < 8; ++r) {
      const int mm = m0 + wm * 16 + rbase + r;
      float v = acc[t][r] + bv;
      if (OP == OP_BIAS_GELU)
        v = 0.5f * v * (1.0f + erff(v * 0.70710678118654752f));
      if (OP == OP_BIAS_RES)
        v += res[(size_t)mm * N + n];
      if (OUTBF) ((unsigned short*)outp)[(size_t)mm * N + n] = bf(v);
      else       ((float*)outp)[(size_t)mm * N + n] = v;
    }
  }
}

// ================= Flash attention on bf16 QKV ==============================
// LDS (halfwords): Ks 32x72 (TDM, pad code 4), Vt 64x40, Ps 8 x 16x40
#define KS_OFF 0
#define VT_OFF (32 * 72)                 // 2304
#define PS_OFF (VT_OFF + 64 * 40)        // 4864
#define AT_HALFS (PS_OFF + 8 * 16 * 40)  // 9984

__global__ __launch_bounds__(256) void attn_fa_wmma(
    const unsigned short* __restrict__ qkv, const unsigned char* __restrict__ mask,
    unsigned short* __restrict__ z)
{
  __shared__ __align__(16) unsigned short sm[AT_HALFS];

  const int tid  = threadIdx.x;
  const int lane = tid & 31;
  const int wave = tid >> 5;
  const int lm   = lane & 15;
  const int hi   = lane >> 4;
  const int c0   = hi ? 16 : 0;
  const int kb16 = hi ? 16 : 0;
  const int h  = blockIdx.y;
  const int b  = blockIdx.z;
  const int q0 = blockIdx.x * 128 + wave * 16;
  const int RS = 3 * D_MODEL;   // qkv row stride (elements)

  // this wave's Q tile (16x64 bf16) straight from global, A-operand layout
  const unsigned short* qrow = qkv + (size_t)(b * SEQ_L + q0 + lm) * RS + h * HEAD_DIM;
  v16bf qa0 = load_aop(qrow,      c0);
  v16bf qa1 = load_aop(qrow + 32, c0);

  v8f acc[4];
  #pragma unroll
  for (int d = 0; d < 4; ++d) acc[d] = (v8f){0.f,0.f,0.f,0.f,0.f,0.f,0.f,0.f};
  float run_max[8], run_sum[8];
  #pragma unroll
  for (int r = 0; r < 8; ++r) { run_max[r] = -1e30f; run_sum[r] = 0.f; }

  for (int kb = 0; kb < SEQ_L / 32; ++kb) {
    __syncthreads();
    // K block (32 keys x 64 dh) via Tensor Data Mover, row pad -> stride 72
    if (wave == 0) {
      const unsigned short* kg =
          qkv + (size_t)(b * SEQ_L + kb * 32) * RS + D_MODEL + h * HEAD_DIM;
      tdm_load_2d(KS_OFF * 2, kg, 64, 32, (unsigned)RS, (unsigned long)RS, 4);
    }
    // V block staged transposed (dh-major) by all threads
    {
      const int key = tid >> 3;
      const int d8  = (tid & 7) * 8;
      const unsigned short* vg =
          qkv + (size_t)(b * SEQ_L + kb * 32 + key) * RS + 2 * D_MODEL
          + h * HEAD_DIM + d8;
      union { uint4 q; unsigned short hw[8]; } u;
      u.q = *(const uint4*)vg;
      #pragma unroll
      for (int j = 0; j < 8; ++j)
        sm[VT_OFF + (d8 + j) * 40 + key] = u.hw[j];
    }
    if (wave == 0) __builtin_amdgcn_s_wait_tensorcnt(0);
    __syncthreads();

    // S = Q @ K^T : two 16x16 tiles, K-dim = dh (2 chunks of 32)
    v8f s[2];
    #pragma unroll
    for (int t = 0; t < 2; ++t) {
      v8f sz = {0.f,0.f,0.f,0.f,0.f,0.f,0.f,0.f};
      v16bf b0 = load_bop(&sm[KS_OFF + (t * 16 + lm) * 72 + 0  + kb16]);
      v16bf b1 = load_bop(&sm[KS_OFF + (t * 16 + lm) * 72 + 32 + kb16]);
      sz = __builtin_amdgcn_wmma_f32_16x16x32_bf16(false, qa0, false, b0,
                                                   (short)0, sz, false, false);
      sz = __builtin_amdgcn_wmma_f32_16x16x32_bf16(false, qa1, false, b1,
                                                   (short)0, sz, false, false);
      #pragma unroll
      for (int r = 0; r < 8; ++r) sz[r] *= 0.125f;   // 1/sqrt(Dh)
      s[t] = sz;
      if (!mask[b * SEQ_L + kb * 32 + t * 16 + lm]) {
        #pragma unroll
        for (int r = 0; r < 8; ++r) s[t][r] = -1e30f;
      }
    }

    // online softmax; rows live per-VGPR, columns per-lane (xor <= 8)
    #pragma unroll
    for (int r = 0; r < 8; ++r) {
      float mx = fmaxf(s[0][r], s[1][r]);
      #pragma unroll
      for (int off = 8; off >= 1; off >>= 1)
        mx = fmaxf(mx, __shfl_xor(mx, off, 32));
      const float nm = fmaxf(run_max[r], mx);
      const float alpha = __expf(run_max[r] - nm);
      run_max[r] = nm;
      const float p0 = __expf(s[0][r] - nm);
      const float p1 = __expf(s[1][r] - nm);
      float ps = p0 + p1;
      #pragma unroll
      for (int off = 8; off >= 1; off >>= 1)
        ps += __shfl_xor(ps, off, 32);
      run_sum[r] = run_sum[r] * alpha + ps;
      #pragma unroll
      for (int d = 0; d < 4; ++d) acc[d][r] *= alpha;
      sm[PS_OFF + wave * 640 + (hi * 8 + r) * 40 + lm]      = bf(p0);
      sm[PS_OFF + wave * 640 + (hi * 8 + r) * 40 + 16 + lm] = bf(p1);
    }
    __syncthreads();

    // P @ V : A = P (16x32), B = V block, 4 n-tiles over dh
    v16bf pa = load_aop(&sm[PS_OFF + wave * 640 + lm * 40], c0);
    #pragma unroll
    for (int d = 0; d < 4; ++d) {
      v16bf vb = load_bop(&sm[VT_OFF + (d * 16 + lm) * 40 + kb16]);
      acc[d] = __builtin_amdgcn_wmma_f32_16x16x32_bf16(false, pa, false, vb,
                                                       (short)0, acc[d], false, false);
    }
  }

  // z[b, q, h*64+dh] (bf16) = acc / run_sum
  const int rbase = hi * 8;
  #pragma unroll
  for (int r = 0; r < 8; ++r) {
    const float inv = 1.0f / run_sum[r];
    const size_t orow = (size_t)(b * SEQ_L + q0 + rbase + r) * D_MODEL
                        + h * HEAD_DIM;
    #pragma unroll
    for (int d = 0; d < 4; ++d)
      z[orow + d * 16 + lm] = bf(acc[d][r] * inv);
  }
}

// ================= host-side launcher =================
extern "C" void kernel_launch(void* const* d_in, const int* in_sizes, int n_in,
                              void* d_out, int out_size, void* d_ws, size_t ws_size,
                              hipStream_t stream)
{
  const float* x      = (const float*)d_in[0];
  const unsigned char* mask = (const unsigned char*)d_in[1];
  const float* ln1_g  = (const float*)d_in[2];
  const float* ln1_b  = (const float*)d_in[3];
  const float* W_qkv  = (const float*)d_in[4];
  const float* b_qkv  = (const float*)d_in[5];
  const float* W_proj = (const float*)d_in[6];
  const float* b_proj = (const float*)d_in[7];
  const float* ln2_g  = (const float*)d_in[8];
  const float* ln2_b  = (const float*)d_in[9];
  const float* W_fc1  = (const float*)d_in[10];
  const float* b_fc1  = (const float*)d_in[11];
  const float* W_fc2  = (const float*)d_in[12];
  const float* b_fc2  = (const float*)d_in[13];
  float* out = (float*)d_out;

  const int M = TOKENS;
  const size_t MB = (size_t)1 << 20;
  char* ws = (char*)d_ws;
  unsigned short* Ybf    = (unsigned short*)(ws + 0);        // 16 MB
  unsigned short* QKVbf  = (unsigned short*)(ws + 16*MB);    // 48 MB
  unsigned short* Zbf    = (unsigned short*)(ws + 64*MB);    // 16 MB
  float*          X1     = (float*)         (ws + 80*MB);    // 32 MB
  unsigned short* HMbf   = (unsigned short*)(ws + 112*MB);   // 64 MB
  unsigned short* WtQKV  = (unsigned short*)(ws + 176*MB);   // 6 MB
  unsigned short* WtPROJ = (unsigned short*)(ws + 182*MB);   // 2 MB
  unsigned short* WtFC1  = (unsigned short*)(ws + 184*MB);   // 8 MB
  unsigned short* WtFC2  = (unsigned short*)(ws + 192*MB);   // 8 MB

  // 0) weights -> bf16, transposed to (N,K)
  wt_transpose_bf16<<<dim3(D_MODEL/64, 3*D_MODEL/64), 256, 0, stream>>>(W_qkv,  WtQKV,  D_MODEL, 3*D_MODEL);
  wt_transpose_bf16<<<dim3(D_MODEL/64,   D_MODEL/64), 256, 0, stream>>>(W_proj, WtPROJ, D_MODEL, D_MODEL);
  wt_transpose_bf16<<<dim3(D_MODEL/64, MLP_HIDDEN/64), 256, 0, stream>>>(W_fc1, WtFC1,  D_MODEL, MLP_HIDDEN);
  wt_transpose_bf16<<<dim3(MLP_HIDDEN/64, D_MODEL/64), 256, 0, stream>>>(W_fc2, WtFC2,  MLP_HIDDEN, D_MODEL);

  // 1) y = LN1(x) -> bf16
  ln_bf16_kernel<<<M, 256, 0, stream>>>(x, ln1_g, ln1_b, Ybf);
  // 2) qkv = y @ W_qkv + b_qkv -> bf16
  gemm_bf16_tdm<OP_BIAS, 1><<<dim3(3*D_MODEL/BN, M/BM), 256, 0, stream>>>(
      Ybf, WtQKV, b_qkv, nullptr, QKVbf, M, 3*D_MODEL, D_MODEL);
  // 3) z = softmax(q k^T / 8 + mask) v -> bf16
  attn_fa_wmma<<<dim3(SEQ_L/128, N_HEADS, SEQ_B), 256, 0, stream>>>(QKVbf, mask, Zbf);
  // 4) x1 = x + z @ W_proj + b_proj -> f32
  gemm_bf16_tdm<OP_BIAS_RES, 0><<<dim3(D_MODEL/BN, M/BM), 256, 0, stream>>>(
      Zbf, WtPROJ, b_proj, x, X1, M, D_MODEL, D_MODEL);
  // 5) y = LN2(x1) -> bf16
  ln_bf16_kernel<<<M, 256, 0, stream>>>(X1, ln2_g, ln2_b, Ybf);
  // 6) h = gelu(y @ W_fc1 + b_fc1) -> bf16
  gemm_bf16_tdm<OP_BIAS_GELU, 1><<<dim3(MLP_HIDDEN/BN, M/BM), 256, 0, stream>>>(
      Ybf, WtFC1, b_fc1, nullptr, HMbf, M, MLP_HIDDEN, D_MODEL);
  // 7) out = x1 + h @ W_fc2 + b_fc2 -> f32
  gemm_bf16_tdm<OP_BIAS_RES, 0><<<dim3(D_MODEL/BN, M/BM), 256, 0, stream>>>(
      HMbf, WtFC2, b_fc2, X1, out, M, D_MODEL, MLP_HIDDEN);
}
